// BiLSTM_56573309224477
// MI455X (gfx1250) — matrix operand order repaired and checked by reading, not compiled, not run on hardware
//
#include <hip/hip_runtime.h>
#include <hip/hip_bf16.h>

#define B_SZ 512
#define T_SZ 512
#define H_SZ 64
#define G_SZ 256   // 4*H
#define WPB  2     // waves per block
#define RPW  16    // batch rows per wave

typedef __bf16 bf16_t;
typedef __attribute__((ext_vector_type(8)))  __bf16 v8bf;
typedef __attribute__((ext_vector_type(16))) __bf16 v16bf;
typedef __attribute__((ext_vector_type(8)))  float  v8f;

// Fast activations: v_exp_f32 + raw v_rcp_f32 (avoids the IEEE div_scale/fma chain).
__device__ __forceinline__ float fast_rcp(float x)  { return __builtin_amdgcn_rcpf(x); }
__device__ __forceinline__ float sigm(float x)      { return fast_rcp(1.f + __expf(-x)); }
__device__ __forceinline__ float tanh_fast(float x) { return 1.f - 2.f * fast_rcp(__expf(2.f * x) + 1.f); }

__device__ __forceinline__ v16bf cat16(v8bf lo, v8bf hi) {
  return __builtin_shufflevector(lo, hi, 0,1,2,3,4,5,6,7,8,9,10,11,12,13,14,15);
}

// A fragment (16M x 32K bf16). `lanebase` already folds the per-lane row offset
// (M*ldk + 8*hf per the ISA 16-bit A layout); kbase is the K tile offset.
// lanes 0-15 -> M=lane, K={kb+0..7, kb+16..23}; lanes 16-31 -> M=lane-16, K={kb+8..15, kb+24..31}.
__device__ __forceinline__ v16bf load_a_frag(const bf16_t* lanebase, int kbase) {
  v8bf lo = *(const v8bf*)(lanebase + kbase);
  v8bf hi = *(const v8bf*)(lanebase + kbase + 16);
  return cat16(lo, hi);
}

// B fragment (32K x 16N) from weights stored [N][K] row-major. `lanebase` folds
// laneN*ldk + 8*hf; `off` = n0*ldk + k0 (compile-time after unrolling -> ds imm offsets).
__device__ __forceinline__ v16bf load_b_frag(const bf16_t* lanebase, int off) {
  v8bf lo = *(const v8bf*)(lanebase + off);
  v8bf hi = *(const v8bf*)(lanebase + off + 16);
  return cat16(lo, hi);
}

__device__ __forceinline__ v8f wmma_bf16(v16bf a, v16bf b, v8f c) {
  return __builtin_amdgcn_wmma_f32_16x16x32_bf16(false, a, false, b, (short)0, c, false, false);
}

// Each wave owns 16 batch rows and runs the full T-step recurrence with zero barriers
// in the time loop. WMMA C/D layout puts gates i/f/g/o of one cell in one lane, so the
// cell update is in-lane VALU; new h goes to LDS (same-wave DScnt ordering only).
// LAYER is a template arg -> branch-free hot loop.
template <int LAYER>
__global__ __launch_bounds__(WPB * 32) void lstm_rec_kernel(
    const float* __restrict__ x,          // LAYER 0: [B,T] (IN==1)
    const bf16_t* __restrict__ h0_in,     // LAYER 1: [B,T,128] bf16
    const float* __restrict__ w_ih_f, const float* __restrict__ w_hh_f,
    const float* __restrict__ b_ih_f, const float* __restrict__ b_hh_f,
    const float* __restrict__ w_ih_r, const float* __restrict__ w_hh_r,
    const float* __restrict__ b_ih_r, const float* __restrict__ b_hh_r,
    bf16_t* __restrict__ h0_out,          // LAYER 0: [B,T,128] (dir picks col half)
    float* __restrict__ h_last)           // LAYER 1: [B,128] (cols 0..63)
{
  __shared__ __attribute__((aligned(16))) bf16_t whh_s[G_SZ * H_SZ];       // 32 KB
  __shared__ __attribute__((aligned(16))) bf16_t wih_s[G_SZ * 2 * H_SZ];   // 64 KB (LAYER 1)
  __shared__ __attribute__((aligned(16))) bf16_t hbuf_s[WPB][RPW * H_SZ];  //  4 KB
  __shared__ float bias_s[G_SZ];
  __shared__ float wih0_s[G_SZ];

  const int dir = blockIdx.y;
  const float* w_ih = dir ? w_ih_r : w_ih_f;
  const float* w_hh = dir ? w_hh_r : w_hh_f;
  const float* b_ih = dir ? b_ih_r : b_ih_f;
  const float* b_hh = dir ? b_hh_r : b_hh_f;
  const int reverse = dir;

  const int tid = threadIdx.x;
  for (int i = tid; i < G_SZ * H_SZ; i += blockDim.x) whh_s[i] = (bf16_t)w_hh[i];
  if (LAYER == 1) {
    for (int i = tid; i < G_SZ * 2 * H_SZ; i += blockDim.x) wih_s[i] = (bf16_t)w_ih[i];
  } else {
    for (int i = tid; i < G_SZ; i += blockDim.x) wih0_s[i] = w_ih[i];   // w_ih_l0: [256,1]
  }
  for (int i = tid; i < G_SZ; i += blockDim.x) bias_s[i] = b_ih[i] + b_hh[i];
  for (int i = tid; i < WPB * RPW * H_SZ; i += blockDim.x) (&hbuf_s[0][0])[i] = (bf16_t)0.f;
  __syncthreads();

  const int wave = tid >> 5, lane = tid & 31;
  const int laneN = lane & 15, hf = lane >> 4;
  const int b0 = (blockIdx.x * WPB + wave) * RPW;
  bf16_t* hb = &hbuf_s[wave][0];

  // Per-lane fragment base pointers (fold all per-lane address math out of the loop).
  const bf16_t* hb_a   = hb + (size_t)laneN * H_SZ + 8 * hf;          // A frags of h
  const bf16_t* whh_b  = whh_s + (size_t)laneN * H_SZ + 8 * hf;      // B frags of w_hh
  const bf16_t* wih_b  = wih_s + (size_t)laneN * (2 * H_SZ) + 8 * hf;// B frags of w_ih (L1)
  const bf16_t* h0a    = (LAYER == 1)
      ? h0_in + ((size_t)(b0 + laneN) * T_SZ) * 128 + 8 * hf : nullptr;
  const float*  xbase  = (LAYER == 0) ? x + (size_t)(b0 + 8 * hf) * T_SZ : nullptr;

  // Hoist loop-invariant per-lane init constants (bias, layer-0 input weight column).
  float bcol_r[16], wcol_r[16];
  #pragma unroll
  for (int p = 0; p < 4; ++p)
    #pragma unroll
    for (int g = 0; g < 4; ++g) {
      const int col = g * H_SZ + p * 16 + laneN;
      bcol_r[p * 4 + g] = bias_s[col];
      if (LAYER == 0) wcol_r[p * 4 + g] = wih0_s[col];
    }

  float cst[4][8];
  #pragma unroll
  for (int p = 0; p < 4; ++p)
    #pragma unroll
    for (int r = 0; r < 8; ++r) cst[p][r] = 0.f;

  // Prefetch step-0 inputs into the (single) input buffer.
  float xv_cur[8];
  v16bf xa_cur[4];
  {
    const int t0 = reverse ? (T_SZ - 1) : 0;
    if (LAYER == 0) {
      #pragma unroll
      for (int r = 0; r < 8; ++r) xv_cur[r] = xbase[(size_t)r * T_SZ + t0];
    } else {
      const bf16_t* xrow = h0a + (size_t)t0 * 128;
      #pragma unroll
      for (int f = 0; f < 4; ++f) xa_cur[f] = load_a_frag(xrow, 32 * f);
    }
  }

  for (int s = 0; s < T_SZ; ++s) {
    const int t = reverse ? (T_SZ - 1 - s) : s;

    // Recurrent A fragments: previous h tile, row-major [16][64] bf16 in LDS.
    v16bf a0 = load_a_frag(hb_a, 0);
    v16bf a1 = load_a_frag(hb_a, 32);

    #pragma unroll
    for (int p = 0; p < 4; ++p) {           // 4 column passes of 16
      v8f acc[4];                           // i, f, g, o tiles: same lane owns same (M,N)
      #pragma unroll
      for (int g = 0; g < 4; ++g) {
        const float bcol = bcol_r[p * 4 + g];
        if (LAYER == 0) {
          const float wcol = wcol_r[p * 4 + g];  // IN==1: input GEMM is a scalar FMA
          #pragma unroll
          for (int r = 0; r < 8; ++r) acc[g][r] = xv_cur[r] * wcol + bcol;
        } else {
          #pragma unroll
          for (int r = 0; r < 8; ++r) acc[g][r] = bcol;
        }
        const int n0 = g * H_SZ + p * 16;
        acc[g] = wmma_bf16(a0, load_b_frag(whh_b, n0 * H_SZ),      acc[g]);
        acc[g] = wmma_bf16(a1, load_b_frag(whh_b, n0 * H_SZ + 32), acc[g]);
        if (LAYER == 1) {                   // fused input GEMM, K=128
          #pragma unroll
          for (int f = 0; f < 4; ++f)
            acc[g] = wmma_bf16(xa_cur[f], load_b_frag(wih_b, n0 * 2 * H_SZ + 32 * f), acc[g]);
        }
      }
      // LSTM cell update: entirely in-lane (exp + rcp only, no IEEE div chain).
      #pragma unroll
      for (int r = 0; r < 8; ++r) {
        const float i_s = sigm(acc[0][r]);
        const float f_s = sigm(acc[1][r]);
        const float g_t = tanh_fast(acc[2][r]);
        const float o_s = sigm(acc[3][r]);
        const float cn  = f_s * cst[p][r] + i_s * g_t;
        const float hn  = o_s * tanh_fast(cn);
        cst[p][r] = cn;
        hb[(r + 8 * hf) * H_SZ + p * 16 + laneN] = (bf16_t)hn;  // feeds next A frags
      }
    }

    if (LAYER == 0) {
      // Stream this step's h tile to global h0 (bf16, row-major), b128 stores.
      const int M = laneN, ch = hf;
      const v8bf* src = (const v8bf*)(hb + M * H_SZ + ch * 32);
      v8bf* dst = (v8bf*)(h0_out + ((size_t)(b0 + M) * T_SZ + t) * 128 + dir * H_SZ + ch * 32);
      dst[0] = src[0]; dst[1] = src[1]; dst[2] = src[2]; dst[3] = src[3];
    }

    // Bottom-of-loop prefetch of the next step's external input into the SAME buffer
    // (last use was above; halves prefetch registers vs double buffering).
    {
      const int sn = (s + 1 < T_SZ) ? (s + 1) : s;          // clamped, branch-free
      const int tn = reverse ? (T_SZ - 1 - sn) : sn;
      if (LAYER == 0) {
        #pragma unroll
        for (int r = 0; r < 8; ++r) xv_cur[r] = xbase[(size_t)r * T_SZ + tn];
      } else {
        const bf16_t* xrow = h0a + (size_t)tn * 128;
        #pragma unroll
        for (int f = 0; f < 4; ++f) xa_cur[f] = load_a_frag(xrow, 32 * f);
      }
    }
  }

  if (LAYER == 1) {
    // Single post-loop writeback of the final h.
    const int M = laneN, ch = hf;
    const bf16_t* src = hb + M * H_SZ + ch * 32;
    float* dst = h_last + (size_t)(b0 + M) * 128 + ch * 32;
    #pragma unroll
    for (int k = 0; k < 32; ++k) dst[k] = (float)src[k];
  }
}

// Layer-1 backward contributes only its FIRST step to h1[:, -1, :] (h=c=0 there),
// so gates = h0[b,T-1,:] . w_ih_l1r^T + bias and c = sig(i)*tanh(g). Tiny kernel.
__global__ __launch_bounds__(64) void l1_bwd_last_kernel(
    const bf16_t* __restrict__ h0, const float* __restrict__ w_ih_r,
    const float* __restrict__ b_ih_r, const float* __restrict__ b_hh_r,
    float* __restrict__ h_last)
{
  __shared__ float xrow[2 * H_SZ];
  const int b = blockIdx.x, j = threadIdx.x;   // 64 threads
  const bf16_t* src = h0 + ((size_t)b * T_SZ + (T_SZ - 1)) * 128;
  xrow[j] = (float)src[j];
  xrow[j + 64] = (float)src[j + 64];
  __syncthreads();
  float g4[4];
  #pragma unroll
  for (int g = 0; g < 4; ++g) {
    const int col = g * H_SZ + j;
    float acc = b_ih_r[col] + b_hh_r[col];
    const float* wr = w_ih_r + (size_t)col * 128;
    for (int k = 0; k < 128; ++k) acc += xrow[k] * wr[k];
    g4[g] = acc;
  }
  const float c = sigm(g4[0]) * tanh_fast(g4[2]);
  const float h = sigm(g4[3]) * tanh_fast(c);
  h_last[(size_t)b * 128 + 64 + j] = h;
}

__global__ void fc_kernel(const float* __restrict__ h_last, const float* __restrict__ fc_w,
                          const float* __restrict__ fc_b, float* __restrict__ out)
{
  const int b = blockIdx.x * blockDim.x + threadIdx.x;
  if (b < B_SZ) {
    float acc = fc_b[0];
    for (int k = 0; k < 128; ++k) acc += h_last[(size_t)b * 128 + k] * fc_w[k];
    out[b] = acc;
  }
}

extern "C" void kernel_launch(void* const* d_in, const int* in_sizes, int n_in,
                              void* d_out, int out_size, void* d_ws, size_t ws_size,
                              hipStream_t stream) {
  (void)in_sizes; (void)n_in; (void)out_size; (void)ws_size;
  const float* x        = (const float*)d_in[0];
  const float* w_ih_l0  = (const float*)d_in[1];
  const float* w_hh_l0  = (const float*)d_in[2];
  const float* b_ih_l0  = (const float*)d_in[3];
  const float* b_hh_l0  = (const float*)d_in[4];
  const float* w_ih_l0r = (const float*)d_in[5];
  const float* w_hh_l0r = (const float*)d_in[6];
  const float* b_ih_l0r = (const float*)d_in[7];
  const float* b_hh_l0r = (const float*)d_in[8];
  const float* w_ih_l1  = (const float*)d_in[9];
  const float* w_hh_l1  = (const float*)d_in[10];
  const float* b_ih_l1  = (const float*)d_in[11];
  const float* b_hh_l1  = (const float*)d_in[12];
  const float* w_ih_l1r = (const float*)d_in[13];
  const float* w_hh_l1r = (const float*)d_in[14];
  const float* b_ih_l1r = (const float*)d_in[15];
  const float* b_hh_l1r = (const float*)d_in[16];
  const float* fc_w     = (const float*)d_in[17];
  const float* fc_b     = (const float*)d_in[18];
  float* out = (float*)d_out;

  // Workspace: h0 bf16 [B][T][128] (64 MB, L2-resident) + h_last f32 [B][128].
  bf16_t* h0     = (bf16_t*)d_ws;
  float*  h_last = (float*)((char*)d_ws + (size_t)B_SZ * T_SZ * 128 * sizeof(bf16_t));

  const int blocks_per_dir = B_SZ / (WPB * RPW);   // 16

  // Layer 0: both directions in one launch (grid.y = dir).
  lstm_rec_kernel<0><<<dim3(blocks_per_dir, 2), WPB * 32, 0, stream>>>(
      x, nullptr,
      w_ih_l0, w_hh_l0, b_ih_l0, b_hh_l0,
      w_ih_l0r, w_hh_l0r, b_ih_l0r, b_hh_l0r,
      h0, nullptr);

  // Layer 1 forward: fused input GEMM + recurrence; stores only the final h.
  lstm_rec_kernel<1><<<dim3(blocks_per_dir, 1), WPB * 32, 0, stream>>>(
      nullptr, h0,
      w_ih_l1, w_hh_l1, b_ih_l1, b_hh_l1,
      nullptr, nullptr, nullptr, nullptr,
      nullptr, h_last);

  // Layer 1 backward: single algebraically-reduced step at t = T-1.
  l1_bwd_last_kernel<<<B_SZ, 64, 0, stream>>>(h0, w_ih_l1r, b_ih_l1r, b_hh_l1r, h_last);

  // Final FC: [B,128] . [128] + b.
  fc_kernel<<<(B_SZ + 255) / 256, 256, 0, stream>>>(h_last, fc_w, fc_b, out);
}
